// MyRnn_3624952398228
// MI455X (gfx1250) — matrix operand order repaired and checked by reading, not compiled, not run on hardware
//
#include <hip/hip_runtime.h>

typedef __attribute__((ext_vector_type(16))) _Float16 v16h;
typedef __attribute__((ext_vector_type(8)))  float    v8f;

#define BATCH   2048
#define SEQ     80
#define EMBED   100
#define UNITS   512
#define KX      128              // embed K padded to multiple of 32
#define BLOCK_M 32               // batch rows per workgroup (2 M-subtiles)
#define NPW     128              // output cols per wave (4 N-slice waves)
#define NT      (NPW / 16)       // 8 wmma n-tiles per wave
#define GRP     4                // n-tiles accumulated per register group
#define LDSH    (UNITS + 8)      // state row stride (halves): 1040B, bank-friendly
#define LDSX    (KX + 8)         // x-buffer row stride (halves)
#define BUF     (BLOCK_M * LDSH) // one state buffer, in halves

#define SMEM_HALVES (6 * BUF + BLOCK_M * LDSX)

// ---- fast transcendental helpers (v_exp_f32 / v_rcp_f32) -------------------
__device__ __forceinline__ float fast_tanh(float x) {
    x = fminf(fmaxf(x, -9.f), 9.f);
    float e = __expf(2.f * x);
    return (e - 1.f) * __builtin_amdgcn_rcpf(e + 1.f);
}
__device__ __forceinline__ float fast_sigmoid(float x) {
    x = fminf(fmaxf(x, -30.f), 30.f);
    return __builtin_amdgcn_rcpf(1.f + __expf(-x));
}

// ---- fragment loaders ------------------------------------------------------
// A (16x32 f16, M x K): lane m (0-15) holds K {k0..k0+7, k0+16..k0+23},
// lane m+16 holds K {k0+8..k0+15, k0+24..k0+31}  (ISA 05_wmma 16-bit A layout)
__device__ __forceinline__ v16h load_a_lds(const _Float16* buf, int stride,
                                           int mbase, int k0, int lrow, int lhi) {
    const _Float16* p = buf + (mbase + lrow) * stride + k0 + lhi * 8;
    v16h a;
    *(uint4*)&a       = *(const uint4*)p;        // 8 halves
    *((uint4*)&a + 1) = *(const uint4*)(p + 16); // 8 halves, +16 halves offset
    return a;
}

// B (32x16 f16, K x N) from W^T stored row-major [N][ldk]:
// lane n (0-15) holds col n K {k0..k0+7, k0+16..k0+23}; lane n+16 holds col n K {k0+8..k0+15, k0+24..k0+31}
__device__ __forceinline__ v16h load_b_glb(const _Float16* Wt, int ldk,
                                           int n0, int k0, int lrow, int lhi) {
    const _Float16* p = Wt + (size_t)(n0 + lrow) * ldk + k0 + lhi * 8;
    v16h b;
    *(uint4*)&b       = *(const uint4*)p;
    *((uint4*)&b + 1) = *(const uint4*)(p + 16);
    return b;
}

// ---- one recurrent layer: snxt = tanh(in @ Wx + scur @ Wh + bias) ----------
// Double-buffered states: no WAR hazard, stores issue per-group immediately.
template<int KIN>
__device__ __forceinline__ void layer_step(const _Float16* __restrict__ inbuf, int in_stride,
                                           const _Float16* __restrict__ scur,
                                           _Float16* __restrict__ snxt,
                                           const _Float16* __restrict__ Wxt,
                                           const _Float16* __restrict__ Wht,
                                           const float* __restrict__ bias,
                                           int mbase, int n0, int lrow, int lhi) {
    // Launder a zero *offset* (not the pointers): keeps kernarg provenance so
    // address-space inference still emits global_load_b128 (flat loads would
    // burn DScnt and the LDS issue path), while making each inlined instance's
    // addresses opaque so LICM cannot hoist/spill the weight stream across the
    // 80-step outer loop.
    size_t zofs = 0;
    asm volatile("" : "+s"(zofs));
    const _Float16* Wx = Wxt + zofs;
    const _Float16* Wh = Wht + zofs;

    #pragma unroll
    for (int g = 0; g < NT / GRP; g++) {
        const int gn0 = n0 + g * (GRP * 16);
        v8f acc[GRP] = {};

        #pragma unroll 1
        for (int k0 = 0; k0 < KIN; k0 += 32) {          // in @ Wx
            v16h a = load_a_lds(inbuf, in_stride, mbase, k0, lrow, lhi);
            #pragma unroll
            for (int j = 0; j < GRP; j++) {
                v16h b = load_b_glb(Wx, KIN, gn0 + j * 16, k0, lrow, lhi);
                acc[j] = __builtin_amdgcn_wmma_f32_16x16x32_f16(
                    false, a, false, b, (short)0, acc[j], false, false);
            }
        }
        #pragma unroll 1
        for (int k0 = 0; k0 < UNITS; k0 += 32) {        // scur @ Wh
            v16h a = load_a_lds(scur, LDSH, mbase, k0, lrow, lhi);
            #pragma unroll
            for (int j = 0; j < GRP; j++) {
                v16h b = load_b_glb(Wh, UNITS, gn0 + j * 16, k0, lrow, lhi);
                acc[j] = __builtin_amdgcn_wmma_f32_16x16x32_f16(
                    false, a, false, b, (short)0, acc[j], false, false);
            }
        }
        // D layout: VGPR r -> (M = r | r+8 by lane half, N = lane&15)
        #pragma unroll
        for (int j = 0; j < GRP; j++) {
            int col  = gn0 + j * 16 + lrow;
            float bv = bias[col];
            #pragma unroll
            for (int r = 0; r < 8; r++) {
                snxt[(mbase + r + lhi * 8) * LDSH + col] =
                    (_Float16)fast_tanh(acc[j][r] + bv);
            }
        }
    }
    __syncthreads();   // all columns of snxt visible to next consumer
}

// ---- fused persistent RNN kernel: one WG owns 32 batch rows for all 80 steps
__global__ __launch_bounds__(256, 1)
void rnn3_fused(const int* __restrict__ tokens, const float* __restrict__ emb,
                const _Float16* __restrict__ Wx0t, const _Float16* __restrict__ Wh0t,
                const float* __restrict__ b0,
                const _Float16* __restrict__ Wx1t, const _Float16* __restrict__ Wh1t,
                const float* __restrict__ b1,
                const _Float16* __restrict__ Wx2t, const _Float16* __restrict__ Wh2t,
                const float* __restrict__ b2,
                const float* __restrict__ Wout, const float* __restrict__ bout,
                float* __restrict__ out) {
    __shared__ _Float16 smem[SMEM_HALVES];
    _Float16* hx = smem + 6 * BUF;

    const int tid   = threadIdx.x;
    const int lane  = tid & 31;
    const int wave  = tid >> 5;           // 0..7
    const int mbase = (wave & 1) * 16;    // 2 M-subtiles
    const int n0    = (wave >> 1) * NPW;  // 4 N-slices of 128 cols
    const int lrow  = lane & 15;
    const int lhi   = lane >> 4;
    const int b0row = blockIdx.x * BLOCK_M;

    {   // zero-init all state buffers (and x-buffer padding)
        uint32_t* p = (uint32_t*)smem;
        for (int i = tid; i < SMEM_HALVES / 2; i += 256) p[i] = 0u;
    }
    __syncthreads();

    for (int t = 0; t < SEQ; t++) {
        // embedding gather -> f16 LDS x-buffer (zero-padded K 100..127)
        {
            int r  = tid >> 3;            // 32 rows, 8 threads/row
            int c0 = (tid & 7) * 16;
            int tok = tokens[(b0row + r) * SEQ + t];
            const float* e = emb + (size_t)tok * EMBED;
            #pragma unroll
            for (int c = 0; c < 16; c++) {
                int cc = c0 + c;
                hx[r * LDSX + cc] = (_Float16)((cc < EMBED) ? e[cc] : 0.f);
            }
        }
        __syncthreads();

        // ping-pong state buffers via pure pointer arithmetic (no movrels)
        const int pc = t & 1;             // current slot
        const int pn = pc ^ 1;            // next slot
        _Float16* s0c = smem + (0 + pc) * BUF;  _Float16* s0n = smem + (0 + pn) * BUF;
        _Float16* s1c = smem + (2 + pc) * BUF;  _Float16* s1n = smem + (2 + pn) * BUF;
        _Float16* s2c = smem + (4 + pc) * BUF;  _Float16* s2n = smem + (4 + pn) * BUF;

        layer_step<KX>   (hx,  LDSX, s0c, s0n, Wx0t, Wh0t, b0, mbase, n0, lrow, lhi);
        layer_step<UNITS>(s0n, LDSH, s1c, s1n, Wx1t, Wh1t, b1, mbase, n0, lrow, lhi);
        layer_step<UNITS>(s1n, LDSH, s2c, s2n, Wx2t, Wh2t, b2, mbase, n0, lrow, lhi);
    }

    // final state: last write went to slot (SEQ-1)&1 ^ 1 == 0 for even SEQ
    const _Float16* s2fin = smem + (4 + (((SEQ - 1) & 1) ^ 1)) * BUF;

    // logits = s2 @ Wout + bout ; sigmoid.  8 lanes per row, 64 elems each.
    {
        int r   = tid >> 3;
        int seg = tid & 7;
        const _Float16* hp = s2fin + r * LDSH + seg * 64;
        const float*    wp = Wout + seg * 64;
        float s = 0.f;
        #pragma unroll 8
        for (int c = 0; c < 64; c++) s += (float)hp[c] * wp[c];
        s += __shfl_xor(s, 1, 32);
        s += __shfl_xor(s, 2, 32);
        s += __shfl_xor(s, 4, 32);
        if (seg == 0) out[b0row + r] = fast_sigmoid(s + bout[0]);
    }
}

// ---- weight pre-pass: f32 [K][N] -> f16 transposed [N][Kpad] ---------------
__global__ void convert_wt(const float* __restrict__ src, _Float16* __restrict__ dst,
                           int Ksrc, int Kpad, int N) {
    int idx = blockIdx.x * blockDim.x + threadIdx.x;
    if (idx >= N * Kpad) return;
    int n = idx / Kpad, k = idx - n * Kpad;
    float v = (k < Ksrc) ? src[(size_t)k * N + n] : 0.f;
    dst[idx] = (_Float16)v;
}

extern "C" void kernel_launch(void* const* d_in, const int* in_sizes, int n_in,
                              void* d_out, int out_size, void* d_ws, size_t ws_size,
                              hipStream_t stream) {
    const int*   tokens = (const int*)  d_in[0];
    const float* emb    = (const float*)d_in[1];
    const float* Wx0 = (const float*)d_in[2];
    const float* Wh0 = (const float*)d_in[3];
    const float* b0  = (const float*)d_in[4];
    const float* Wx1 = (const float*)d_in[5];
    const float* Wh1 = (const float*)d_in[6];
    const float* b1  = (const float*)d_in[7];
    const float* Wx2 = (const float*)d_in[8];
    const float* Wh2 = (const float*)d_in[9];
    const float* b2  = (const float*)d_in[10];
    const float* Wout = (const float*)d_in[11];
    const float* bout = (const float*)d_in[12];
    float* out = (float*)d_out;

    _Float16* ws   = (_Float16*)d_ws;
    _Float16* Wx0t = ws;                       // 512 x 128
    _Float16* Wh0t = Wx0t + 512 * 128;         // 512 x 512
    _Float16* Wx1t = Wh0t + 512 * 512;
    _Float16* Wh1t = Wx1t + 512 * 512;
    _Float16* Wx2t = Wh1t + 512 * 512;
    _Float16* Wh2t = Wx2t + 512 * 512;

    const int T = 256;
    convert_wt<<<(512 * 128 + T - 1) / T, T, 0, stream>>>(Wx0, Wx0t, EMBED, KX, UNITS);
    convert_wt<<<(512 * 512 + T - 1) / T, T, 0, stream>>>(Wh0, Wh0t, UNITS, UNITS, UNITS);
    convert_wt<<<(512 * 512 + T - 1) / T, T, 0, stream>>>(Wx1, Wx1t, UNITS, UNITS, UNITS);
    convert_wt<<<(512 * 512 + T - 1) / T, T, 0, stream>>>(Wh1, Wh1t, UNITS, UNITS, UNITS);
    convert_wt<<<(512 * 512 + T - 1) / T, T, 0, stream>>>(Wx2, Wx2t, UNITS, UNITS, UNITS);
    convert_wt<<<(512 * 512 + T - 1) / T, T, 0, stream>>>(Wh2, Wh2t, UNITS, UNITS, UNITS);

    rnn3_fused<<<BATCH / BLOCK_M, 256, 0, stream>>>(
        tokens, emb, Wx0t, Wh0t, b0, Wx1t, Wh1t, b1, Wx2t, Wh2t, b2,
        Wout, bout, out);
}
